// NodeEmbeddingID_70738111365650
// MI455X (gfx1250) — matrix operand order repaired and checked by reading, not compiled, not run on hardware
//
#include <hip/hip_runtime.h>

// ---------------------------------------------------------------------------
// CDNA5 (gfx1250) implementation of the TGN-style reference.
//
// Exact simplification: reference softmax is over a singleton axis, so
// attention weights == 1.0 and attn(q,c) == (c @ Wv^T + b_v) @ Wo^T + bo.
// Wq / Wk / the query path cannot affect the output.
//
// GEMMs run on v_wmma_f32_16x16x32_bf16 (wave32, 16x16 f32 accumulator in
// 8 VGPRs). Scatter-aggregation uses global_atomic_add_f32; per-dst time max
// uses bit-pattern u32 atomicMax (exact for non-negative floats; the cnt>0
// guard is subsumed by zero-init since te >= 0).
//
// Bounds handling: out-of-range tile rows are CLAMPED (not branch-guarded) in
// the A loaders — a clamped row only pollutes its own accumulator row / LDS
// row, which is never stored (m < N at the store). This keeps the load path
// branch-free so the compiler emits clause-batched global_load_b128.
// ---------------------------------------------------------------------------

typedef __attribute__((ext_vector_type(16))) __bf16 v16bf;
typedef __attribute__((ext_vector_type(8)))  float  v8f;

__device__ __forceinline__ v8f wmma_bf16(v16bf a, v16bf b, v8f c) {
  // D = A(16x32 bf16) x B(32x16 bf16) + C(16x16 f32)
  return __builtin_amdgcn_wmma_f32_16x16x32_bf16(
      /*neg_a=*/false, a, /*neg_b=*/false, b,
      /*c_mod=*/(short)0, c, /*reuse_a=*/false, /*reuse_b=*/false);
}

// A-matrix 16x32 bf16 fragment from a row-major fp32 matrix.
// ISA layout: lane m = lane&15 holds row row0+m; half h = lane>>4 selects the
// K sub-block. Per lane the 16 elements are two contiguous 8-float runs:
//   run0: cols [col0 + 8h,      +7]   (VGPRs 0..3)
//   run1: cols [col0 + 16 + 8h, +7]   (VGPRs 4..7)
__device__ __forceinline__ v16bf load_A_f32(const float* M, int row0, int nrows,
                                            int col0, int ld) {
  int lane = threadIdx.x & 31;
  int m = lane & 15;
  int h = lane >> 4;
  int r = row0 + m;
  r = (r < nrows) ? r : (nrows - 1);  // clamp: garbage rows never stored
  const float* base = M + (size_t)r * ld + col0 + 8 * h;
  float4 x0 = ((const float4*)base)[0];
  float4 x1 = ((const float4*)base)[1];
  float4 y0 = ((const float4*)(base + 16))[0];
  float4 y1 = ((const float4*)(base + 16))[1];
  v16bf a;
  a[0]  = (__bf16)x0.x; a[1]  = (__bf16)x0.y; a[2]  = (__bf16)x0.z; a[3]  = (__bf16)x0.w;
  a[4]  = (__bf16)x1.x; a[5]  = (__bf16)x1.y; a[6]  = (__bf16)x1.z; a[7]  = (__bf16)x1.w;
  a[8]  = (__bf16)y0.x; a[9]  = (__bf16)y0.y; a[10] = (__bf16)y0.z; a[11] = (__bf16)y0.w;
  a[12] = (__bf16)y1.x; a[13] = (__bf16)y1.y; a[14] = (__bf16)y1.z; a[15] = (__bf16)y1.w;
  return a;
}

// Same A layout sourced from a bf16 LDS tile (rows always valid).
__device__ __forceinline__ v16bf load_A_lds(const __bf16* L, int col0, int ld) {
  int lane = threadIdx.x & 31;
  int m = lane & 15;
  int h = lane >> 4;
  v16bf a;
  const __bf16* p0 = L + m * ld + col0 + 8 * h;
  const __bf16* p1 = p0 + 16;
#pragma unroll
  for (int v = 0; v < 8; ++v) a[v] = p0[v];
#pragma unroll
  for (int v = 0; v < 8; ++v) a[8 + v] = p1[v];
  return a;
}

// B-matrix 32x16 bf16 fragment for B = W^T, W row-major [nOut][K].
// Lane holds column n = n0 + (lane&15) == row n of W; per lane the 16
// elements are K = k0 + 8h .. +7 and k0 + 16 + 8h .. +7 (two 8-float runs).
__device__ __forceinline__ v16bf load_B_WT(const float* W, int n0, int k0, int ldw) {
  int lane = threadIdx.x & 31;
  int n = n0 + (lane & 15);
  int h = lane >> 4;
  const float* base = W + (size_t)n * ldw + k0 + 8 * h;
  float4 x0 = ((const float4*)base)[0];
  float4 x1 = ((const float4*)base)[1];
  float4 x2 = ((const float4*)(base + 16))[0];
  float4 x3 = ((const float4*)(base + 16))[1];
  v16bf b;
  b[0]  = (__bf16)x0.x; b[1]  = (__bf16)x0.y; b[2]  = (__bf16)x0.z; b[3]  = (__bf16)x0.w;
  b[4]  = (__bf16)x1.x; b[5]  = (__bf16)x1.y; b[6]  = (__bf16)x1.z; b[7]  = (__bf16)x1.w;
  b[8]  = (__bf16)x2.x; b[9]  = (__bf16)x2.y; b[10] = (__bf16)x2.z; b[11] = (__bf16)x2.w;
  b[12] = (__bf16)x3.x; b[13] = (__bf16)x3.y; b[14] = (__bf16)x3.z; b[15] = (__bf16)x3.w;
  return b;
}

// ---------------------------------------------------------------------------
// Kernel 1: h = relu(V @ W^T + bias) + S      (V:[N,128], W:[128,128])
// 8 waves per block; wave w owns the 16x16 output tile at column w*16.
// ---------------------------------------------------------------------------
__global__ __launch_bounds__(256) void k_decompose(
    const float* __restrict__ V, const float* __restrict__ W,
    const float* __restrict__ bias, const float* __restrict__ S,
    float* __restrict__ H, int N) {
  int wave = threadIdx.x >> 5;
  int lane = threadIdx.x & 31;
  int row0 = blockIdx.x * 16;
  int col0 = wave * 16;

  v8f acc = {};
#pragma unroll
  for (int k0 = 0; k0 < 128; k0 += 32) {
    v16bf a = load_A_f32(V, row0, N, k0, 128);
    v16bf b = load_B_WT(W, col0, k0, 128);
    acc = wmma_bf16(a, b, acc);
  }
  int n = col0 + (lane & 15);
  int hh = (lane >> 4) * 8;
  float bn = bias[n];
#pragma unroll
  for (int r = 0; r < 8; ++r) {
    int m = row0 + r + hh;
    if (m < N) {
      float v = fmaxf(acc[r] + bn, 0.f) + S[(size_t)m * 128 + n];
      H[(size_t)m * 128 + n] = v;
    }
  }
}

// ---------------------------------------------------------------------------
// Kernel 2: per-dst max of edge timestamps (exact via u32 bits: te in [0,1)).
// ---------------------------------------------------------------------------
__global__ __launch_bounds__(256) void k_t0max(
    const float* __restrict__ te, const int* __restrict__ dst,
    float* __restrict__ t0, int E) {
  int i = blockIdx.x * blockDim.x + threadIdx.x;
  if (i < E)
    atomicMax((unsigned int*)(t0 + dst[i]), __float_as_uint(te[i]));
}

// ---------------------------------------------------------------------------
// Kernel 3: message scatter. One wave per edge:
//   C[dst] += concat(Hsrc[src], Ef[edge], cos((t0[dst]-te)*freq))   (384 wide)
// freq[k] = 10^(-9k/127) = exp(-k * 9*ln(10)/127)
// ---------------------------------------------------------------------------
__global__ __launch_bounds__(256) void k_scatter(
    const float* __restrict__ Hsrc, const float* __restrict__ Ef,
    const float* __restrict__ te, const int* __restrict__ src,
    const int* __restrict__ dst, const float* __restrict__ t0,
    float* __restrict__ C, int E) {
  int edge = blockIdx.x * 8 + (threadIdx.x >> 5);
  if (edge >= E) return;
  int lane = threadIdx.x & 31;
  int s = src[edge];
  int d = dst[edge];
  float dt = t0[d] - te[edge];
  const float* hs = Hsrc + (size_t)s * 128;
  const float* ef = Ef + (size_t)edge * 128;
  float* c = C + (size_t)d * 384;
  const float kLnF = -0.16317532074f;  // -9*ln(10)/127
#pragma unroll
  for (int j = 0; j < 4; ++j) {
    int col = lane + j * 32;
    atomicAdd(&c[col], hs[col]);
    atomicAdd(&c[col + 128], ef[col]);
    float f = __expf(kLnF * (float)col);
    atomicAdd(&c[col + 256], __cosf(dt * f));
  }
}

// ---------------------------------------------------------------------------
// Kernel 4: fused attention (softmax==1):
//   vp = C @ Wv^T + bv     [16 x 256]  (staged bf16 in LDS)
//   H2 = vp @ Wo^T + bo    [16 x 256]
// 8 waves per block, 16 rows per block; each wave does 2 column tiles.
// ---------------------------------------------------------------------------
__global__ __launch_bounds__(256) void k_attn(
    const float* __restrict__ C, const float* __restrict__ Wv,
    const float* __restrict__ Wo, const float* __restrict__ bv,
    const float* __restrict__ bo, float* __restrict__ H2, int N) {
  __shared__ __bf16 vp[16][256 + 8];
  int wave = threadIdx.x >> 5;
  int lane = threadIdx.x & 31;
  int row0 = blockIdx.x * 16;
  int hh = (lane >> 4) * 8;

#pragma unroll
  for (int tt = 0; tt < 2; ++tt) {
    int col0 = (wave + tt * 8) * 16;
    v8f acc = {};
#pragma unroll
    for (int k0 = 0; k0 < 384; k0 += 32) {
      v16bf a = load_A_f32(C, row0, N, k0, 384);
      v16bf b = load_B_WT(Wv, col0, k0, 384);
      acc = wmma_bf16(a, b, acc);
    }
    int n = col0 + (lane & 15);
    float bn = bv[n];
#pragma unroll
    for (int r = 0; r < 8; ++r) vp[r + hh][n] = (__bf16)(acc[r] + bn);
  }
  __syncthreads();

#pragma unroll
  for (int tt = 0; tt < 2; ++tt) {
    int col0 = (wave + tt * 8) * 16;
    v8f acc = {};
#pragma unroll
    for (int k0 = 0; k0 < 256; k0 += 32) {
      v16bf a = load_A_lds(&vp[0][0], k0, 256 + 8);
      v16bf b = load_B_WT(Wo, col0, k0, 256);
      acc = wmma_bf16(a, b, acc);
    }
    int n = col0 + (lane & 15);
    float bn = bo[n];
#pragma unroll
    for (int r = 0; r < 8; ++r) {
      int m = row0 + r + hh;
      if (m < N) H2[(size_t)m * 256 + n] = acc[r] + bn;
    }
  }
}

// ---------------------------------------------------------------------------
// Kernel 5: fused head:
//   x   = relu(concat(H[128], H2[256]) @ W1^T + b1)   [16 x 128] (bf16 LDS)
//   out = x @ W2^T + b2                               [16 x 128]
// ---------------------------------------------------------------------------
__global__ __launch_bounds__(256) void k_head(
    const float* __restrict__ H, const float* __restrict__ H2,
    const float* __restrict__ W1, const float* __restrict__ b1,
    const float* __restrict__ W2, const float* __restrict__ b2,
    float* __restrict__ Out, int N) {
  __shared__ __bf16 x[16][128 + 8];
  int wave = threadIdx.x >> 5;
  int lane = threadIdx.x & 31;
  int row0 = blockIdx.x * 16;
  int col0 = wave * 16;
  int hh = (lane >> 4) * 8;
  int n = col0 + (lane & 15);

  v8f acc = {};
#pragma unroll
  for (int k0 = 0; k0 < 128; k0 += 32) {  // K cols 0..127 <- H
    v16bf a = load_A_f32(H, row0, N, k0, 128);
    v16bf b = load_B_WT(W1, col0, k0, 384);
    acc = wmma_bf16(a, b, acc);
  }
#pragma unroll
  for (int k0 = 0; k0 < 256; k0 += 32) {  // K cols 128..383 <- H2
    v16bf a = load_A_f32(H2, row0, N, k0, 256);
    v16bf b = load_B_WT(W1, col0, 128 + k0, 384);
    acc = wmma_bf16(a, b, acc);
  }
  float bn = b1[n];
#pragma unroll
  for (int r = 0; r < 8; ++r)
    x[r + hh][n] = (__bf16)fmaxf(acc[r] + bn, 0.f);
  __syncthreads();

  v8f acc2 = {};
#pragma unroll
  for (int k0 = 0; k0 < 128; k0 += 32) {
    v16bf a = load_A_lds(&x[0][0], k0, 128 + 8);
    v16bf b = load_B_WT(W2, col0, k0, 128);
    acc2 = wmma_bf16(a, b, acc2);
  }
  bn = b2[n];
#pragma unroll
  for (int r = 0; r < 8; ++r) {
    int m = row0 + r + hh;
    if (m < N) Out[(size_t)m * 128 + n] = acc2[r] + bn;
  }
}

// ---------------------------------------------------------------------------
// Host-side launch
// ---------------------------------------------------------------------------
extern "C" void kernel_launch(void* const* d_in, const int* in_sizes, int n_in,
                              void* d_out, int out_size, void* d_ws, size_t ws_size,
                              hipStream_t stream) {
  // Inputs per setup_inputs() order
  const float* vi   = (const float*)d_in[0];
  const float* vj   = (const float*)d_in[1];
  const float* vi_r = (const float*)d_in[2];
  const float* vj_r = (const float*)d_in[3];
  const float* vi_n = (const float*)d_in[4];
  const float* vj_n = (const float*)d_in[5];
  const float* si   = (const float*)d_in[6];
  const float* sj   = (const float*)d_in[7];
  const float* si_r = (const float*)d_in[8];
  const float* sj_r = (const float*)d_in[9];
  const float* si_n = (const float*)d_in[10];
  const float* sj_n = (const float*)d_in[11];
  const float* e    = (const float*)d_in[12];
  const float* e_r  = (const float*)d_in[13];
  const float* e_n  = (const float*)d_in[14];
  const float* t    = (const float*)d_in[15];
  const float* t_r  = (const float*)d_in[16];
  const float* t_n  = (const float*)d_in[17];
  const float* Wdu  = (const float*)d_in[18];
  const float* bdu  = (const float*)d_in[19];
  const float* Wdv  = (const float*)d_in[20];
  const float* bdv  = (const float*)d_in[21];
  const float* W1   = (const float*)d_in[22];
  const float* b1   = (const float*)d_in[23];
  const float* W2   = (const float*)d_in[24];
  const float* b2   = (const float*)d_in[25];
  // d_in[26] Wq, d_in[27] Wk, b_in[0:512]: dead (softmax over singleton == 1)
  const float* Wv   = (const float*)d_in[28];
  const float* b_in = (const float*)d_in[29];
  const float* Wo   = (const float*)d_in[30];
  const float* bo   = (const float*)d_in[31];
  const int* src_g  = (const int*)d_in[32];
  const int* dst_g  = (const int*)d_in[33];
  const int* src_r  = (const int*)d_in[34];
  const int* dst_r  = (const int*)d_in[35];
  const int* src_n  = (const int*)d_in[36];
  const int* dst_n  = (const int*)d_in[37];

  const int N = in_sizes[0] / 128;   // 12500
  const int E = in_sizes[15];        // 200000
  const float* bv = b_in + 512;      // value-projection bias

  // Workspace layout (floats)
  float* ws  = (float*)d_ws;
  float* c0  = ws;                       // [N,384] cj
  float* c1  = c0 + (size_t)N * 384;     // [N,384] ci
  float* c2  = c1 + (size_t)N * 384;     // [N,384] cn
  float* t0g = c2 + (size_t)N * 384;     // [N] x3 (t0 for g, r, n)
  float* t0r = t0g + N;
  float* t0n = t0r + N;
  float* h   = t0n + N;                  // 6 x [N,128]: hi,hj,hi_r,hj_r,hi_n,hj_n
  float* hi    = h;
  float* hj    = h + (size_t)1 * N * 128;
  float* hi_rp = h + (size_t)2 * N * 128;
  float* hj_rp = h + (size_t)3 * N * 128;
  float* hi_np = h + (size_t)4 * N * 128;
  float* hj_np = h + (size_t)5 * N * 128;
  float* h2  = h + (size_t)6 * N * 128;  // 3 x [N,256]: hj2, hi2, hn2
  float* hj2 = h2;
  float* hi2 = h2 + (size_t)N * 256;
  float* hn2 = h2 + (size_t)2 * N * 256;

  float* out0 = (float*)d_out;                       // head(hj_r, hi2)
  float* out1 = out0 + (size_t)N * 128;              // head(hj,   hj2)
  float* out2 = out0 + (size_t)2 * N * 128;          // head(hj_n, hn2)

  // Zero the accumulators (c arrays + t0 arrays are contiguous at ws base)
  hipMemsetAsync(ws, 0, ((size_t)3 * N * 384 + 3 * N) * sizeof(float), stream);

  dim3 blk(256);
  int gTiles = (N + 15) / 16;

  // Stage A: node decompositions
  k_decompose<<<gTiles, blk, 0, stream>>>(vi,   Wdu, bdu, si,   hi,    N);
  k_decompose<<<gTiles, blk, 0, stream>>>(vj,   Wdv, bdv, sj,   hj,    N);
  k_decompose<<<gTiles, blk, 0, stream>>>(vi_r, Wdv, bdv, si_r, hi_rp, N);
  k_decompose<<<gTiles, blk, 0, stream>>>(vj_r, Wdu, bdu, sj_r, hj_rp, N);
  k_decompose<<<gTiles, blk, 0, stream>>>(vi_n, Wdu, bdu, si_n, hi_np, N);
  k_decompose<<<gTiles, blk, 0, stream>>>(vj_n, Wdv, bdv, sj_n, hj_np, N);

  // Stage B: per-destination timestamp max
  int gE = (E + 255) / 256;
  k_t0max<<<gE, blk, 0, stream>>>(t,   dst_g, t0g, E);
  k_t0max<<<gE, blk, 0, stream>>>(t_r, dst_r, t0r, E);
  k_t0max<<<gE, blk, 0, stream>>>(t_n, dst_n, t0n, E);

  // Stage C: message scatter-sum (one wave per edge)
  int gS = (E + 7) / 8;
  k_scatter<<<gS, blk, 0, stream>>>(hi,    e,   t,   src_g, dst_g, t0g, c0, E);
  k_scatter<<<gS, blk, 0, stream>>>(hi_rp, e_r, t_r, src_r, dst_r, t0r, c1, E);
  k_scatter<<<gS, blk, 0, stream>>>(hi_np, e_n, t_n, src_n, dst_n, t0n, c2, E);

  // Stage D: attention (value+output projection only; softmax == 1)
  k_attn<<<gTiles, blk, 0, stream>>>(c0, Wv, Wo, bv, bo, hj2, N);
  k_attn<<<gTiles, blk, 0, stream>>>(c1, Wv, Wo, bv, bo, hi2, N);
  k_attn<<<gTiles, blk, 0, stream>>>(c2, Wv, Wo, bv, bo, hn2, N);

  // Stage E: output heads (reference return order)
  k_head<<<gTiles, blk, 0, stream>>>(hj_rp, hi2, W1, b1, W2, b2, out0, N);
  k_head<<<gTiles, blk, 0, stream>>>(hj,    hj2, W1, b1, W2, b2, out1, N);
  k_head<<<gTiles, blk, 0, stream>>>(hj_np, hn2, W1, b1, W2, b2, out2, N);
}
